// ConvOffset2D_nonlocal2_1614907703718
// MI455X (gfx1250) — matrix operand order repaired and checked by reading, not compiled, not run on hardware
//
#include <hip/hip_runtime.h>
#include <stdint.h>

// ---------------------------------------------------------------------------
// Problem constants
// ---------------------------------------------------------------------------
constexpr int Bn = 16, Cn = 128, Jn = 25, Tn = 256;
constexpr int JT   = Jn * Tn;      // 6400 pixels per image
constexpr int NPIX = Bn * JT;      // 102400 pixels total (BN population per channel)

typedef __attribute__((ext_vector_type(16))) _Float16     v16h;
typedef __attribute__((ext_vector_type(8)))  float        v8f;
typedef __attribute__((ext_vector_type(4)))  unsigned int u32x4;
typedef __attribute__((ext_vector_type(8)))  int          i32x8;
typedef __attribute__((ext_vector_type(4)))  int          i32x4;
typedef __attribute__((ext_vector_type(4)))  unsigned int u4v;   // 16-byte chunk

union V16H { u4v u[2]; v16h v; };

// ---------------------------------------------------------------------------
// Tensor Data Mover: 2D tile load global->LDS (CDNA5 ISA ch.8 descriptor).
// This toolchain exposes the 6-arg builtin:
//   (u32x4 g0, i32x8 g1, i32x4 g2, i32x4 g3, i32x8 g4, i32 cpol)
// Group0: count=1 | lds_addr | global_addr[56:0] | type=2.
// Group1: ds/pad | tensor_dim0/1 | tile_dim0/1 | tensor_dim0_stride.
// Groups 2/3/4 zero (2D tile: tile_dim2 == 0 -> higher dims unused).
// ---------------------------------------------------------------------------
__device__ inline void tdm_load_2d(uint32_t lds_off, const void* gptr,
                                   uint32_t dsize_code,          // 0=1B,1=2B,2=4B
                                   uint32_t tdim0, uint32_t tdim1,
                                   uint64_t stride0,             // elements
                                   uint32_t tile0, uint32_t tile1,
                                   uint32_t pad_en, uint32_t pad_int_code,
                                   uint32_t pad_amt_code)
{
  uint64_t ga = (uint64_t)(uintptr_t)gptr;
  u32x4 g0;
  g0[0] = 1u;                                   // one valid user descriptor
  g0[1] = lds_off;                              // LDS byte offset (dynamic smem base)
  g0[2] = (uint32_t)ga;                         // global_addr[31:0]
  g0[3] = (uint32_t)((ga >> 32) & 0x01FFFFFFull) | (2u << 30);  // addr[56:32] | type=2
  uint32_t w0 = (dsize_code << 16) | (pad_en << 20) |
                (pad_int_code << 22) | (pad_amt_code << 25);    // wg_mask=0
  uint32_t w1 = (tdim0 & 0xFFFFu) << 16;                        // tensor_dim0[15:0]
  uint32_t w2 = ((tdim0 >> 16) & 0xFFFFu) | ((tdim1 & 0xFFFFu) << 16);
  uint32_t w3 = ((tdim1 >> 16) & 0xFFFFu) | ((tile0 & 0xFFFFu) << 16);
  uint32_t w4 = (tile1 & 0xFFFFu);                              // tile_dim2 = 0 (2D)
  uint32_t w5 = (uint32_t)stride0;
  uint32_t w6 = (uint32_t)((stride0 >> 32) & 0xFFFFull);        // dim1_stride unused
  uint32_t w7 = 0;
  i32x8 g1 = { (int)w0, (int)w1, (int)w2, (int)w3,
               (int)w4, (int)w5, (int)w6, (int)w7 };
  i32x4 gz  = { 0, 0, 0, 0 };
  i32x8 gz8 = { 0, 0, 0, 0, 0, 0, 0, 0 };
  __builtin_amdgcn_tensor_load_to_lds(g0, g1, gz, gz, gz8, 0);
}

// ---------------------------------------------------------------------------
// Prep: transpose-convert x (B,C,J,T) f32 -> xhT[b][q][c] f16 (q = j*T+t).
// [q][c] layout makes WMMA B-fragments single contiguous 32B LDS reads.
// ---------------------------------------------------------------------------
__global__ void __launch_bounds__(256) prep_x(const float* __restrict__ x,
                                              _Float16* __restrict__ xhT)
{
  __shared__ float sh[16][17];
  int ct = blockIdx.x & 7;             // 8 channel tiles
  int qt = (blockIdx.x >> 3) % 400;    // 400 q tiles
  int b  = blockIdx.x / 3200;
  int tx = threadIdx.x & 15, ty = threadIdx.x >> 4;
  int cin = ct * 16 + ty, qin = qt * 16 + tx;
  sh[ty][tx] = x[((size_t)(b * Cn + cin)) * JT + qin];
  __syncthreads();
  int qo = qt * 16 + ty, co = ct * 16 + tx;
  xhT[((size_t)(b * JT + qo)) * Cn + co] = (_Float16)sh[tx][ty];
}

__global__ void __launch_bounds__(256) prep_w(const float* __restrict__ wI,
                                              const float* __restrict__ wE,
                                              _Float16* __restrict__ WI,
                                              _Float16* __restrict__ WE)
{
  int i = blockIdx.x * 256 + threadIdx.x;
  if (i < Cn * Cn) { WI[i] = (_Float16)wI[i]; WE[i] = (_Float16)wE[i]; }
}

// ---------------------------------------------------------------------------
// WMMA conv1x1 (both branches), relu fused.
// MODE 0: accumulate per-channel sum/sumsq for BN batch stats.
// MODE 1: a_int/a_ext = sum_c wfold[c]*relu(conv+bias) + c0 (BN folded).
// Block: 256 thr (8 waves); block tile = 128 channels x 128 pixels; K=128.
// x tile DMA'd into LDS by TDM, rows padded 256B->272B (pad 8 halves).
// ---------------------------------------------------------------------------
template <int MODE>
__global__ void __launch_bounds__(256) conv_wmma_kernel(
    const _Float16* __restrict__ xhT,
    const _Float16* __restrict__ WI, const _Float16* __restrict__ WE,
    const float* __restrict__ bI, const float* __restrict__ bE,
    float* __restrict__ statSumI, float* __restrict__ statSqI,
    float* __restrict__ statSumE, float* __restrict__ statSqE,
    const float* __restrict__ wfI, const float* __restrict__ wfE,
    const float* __restrict__ c0,
    float* __restrict__ aIg, float* __restrict__ aEg)
{
  extern __shared__ char smemraw[];
  _Float16* xs  = (_Float16*)smemraw;            // 128 rows x 136 halves (272B)
  float*   ldsA = (float*)(smemraw + 34816);     // MODE 1: aI[128] | aE[128]

  const int tid  = threadIdx.x;
  const int lane = tid & 31, wv = tid >> 5;
  const int hl   = lane >> 4;        // half-wave select
  const int lm   = lane & 15;
  const int ow   = wv * 16;          // this wave's channel base
  const int b    = blockIdx.x / 50;
  const int qbase = (blockIdx.x % 50) * 128;

  if (wv == 0) {
    // f16 tensor: dim0=c(128, contiguous), dim1=q; tile 128x128, pad 64DW+4DW
    tdm_load_2d(0, xhT + ((size_t)(b * JT + qbase)) * Cn,
                /*dsize*/1, /*tdim0*/128, /*tdim1*/128, /*stride0*/128,
                /*tile0*/128, /*tile1*/128, /*pad*/1, /*64 DW*/5, /*4 DW*/3);
  }
  if (MODE == 1 && tid < 128) { ldsA[tid] = 0.f; ldsA[128 + tid] = 0.f; }
  if (wv == 0) __builtin_amdgcn_s_wait_tensorcnt(0);
  __syncthreads();

  float sI[8] = {0}, qqI[8] = {0}, sE[8] = {0}, qqE[8] = {0};
  float wfIv[8], wfEv[8], bIv[8], bEv[8];
#pragma unroll
  for (int r = 0; r < 8; r++) {
    int ch = ow + 8 * hl + r;        // C/D layout: M = r + 8*(lane>=16)
    bIv[r] = bI[ch]; bEv[r] = bE[ch];
    if (MODE == 1) { wfIv[r] = wfI[ch]; wfEv[r] = wfE[ch]; }
  }

  for (int n = 0; n < 8; n++) {      // 8 output tiles of 16 pixels per wave
    v8f accI = {0.f,0.f,0.f,0.f,0.f,0.f,0.f,0.f};
    v8f accE = {0.f,0.f,0.f,0.f,0.f,0.f,0.f,0.f};
    const int qrow = n * 16 + lm;
#pragma unroll
    for (int k4 = 0; k4 < 4; k4++) {
      const int k = k4 * 32;
      // B frag (32x16, f16): lanes 0-15 K=0..15, lanes 16-31 K=16..31
      V16H bf;
      const u4v* bp = (const u4v*)(xs + (size_t)qrow * 136 + k + 16 * hl);
      bf.u[0] = bp[0]; bf.u[1] = bp[1];
      // A frags (16x32, f16): row=o, lanes 0-15 K=0..15, lanes 16-31 K=16..31
      const int arow = ow + lm;
      const int ka = k + 16 * hl;
      V16H afI, afE;
      afI.u[0] = *(const u4v*)(WI + (size_t)arow * Cn + ka);
      afI.u[1] = *(const u4v*)(WI + (size_t)arow * Cn + ka + 8);
      afE.u[0] = *(const u4v*)(WE + (size_t)arow * Cn + ka);
      afE.u[1] = *(const u4v*)(WE + (size_t)arow * Cn + ka + 8);
      accI = __builtin_amdgcn_wmma_f32_16x16x32_f16(false, afI.v, false, bf.v,
                                                    (short)0, accI, false, false);
      accE = __builtin_amdgcn_wmma_f32_16x16x32_f16(false, afE.v, false, bf.v,
                                                    (short)0, accE, false, false);
    }
    // Epilogue: bias + relu, then stats (MODE 0) or w_fus dot (MODE 1)
    float dI = 0.f, dE = 0.f;
#pragma unroll
    for (int r = 0; r < 8; r++) {
      float vI = fmaxf(accI[r] + bIv[r], 0.f);
      float vE = fmaxf(accE[r] + bEv[r], 0.f);
      if (MODE == 0) {
        sI[r] += vI; qqI[r] += vI * vI;
        sE[r] += vE; qqE[r] += vE * vE;
      } else {
        dI += wfIv[r] * vI; dE += wfEv[r] * vE;
      }
    }
    if (MODE == 1) {
      atomicAdd(&ldsA[qrow], dI);        // column q accumulates over channels
      atomicAdd(&ldsA[128 + qrow], dE);
    }
  }

  if (MODE == 0) {
    // reduce each channel's 16 q-lanes, then one atomic per channel
#pragma unroll
    for (int r = 0; r < 8; r++) {
      float a = sI[r], b2 = qqI[r], c2 = sE[r], d2 = qqE[r];
#pragma unroll
      for (int m = 1; m < 16; m <<= 1) {
        a  += __shfl_xor(a,  m, 32); b2 += __shfl_xor(b2, m, 32);
        c2 += __shfl_xor(c2, m, 32); d2 += __shfl_xor(d2, m, 32);
      }
      if (lm == 0) {
        int ch = ow + 8 * hl + r;
        atomicAdd(&statSumI[ch], a);  atomicAdd(&statSqI[ch], b2);
        atomicAdd(&statSumE[ch], c2); atomicAdd(&statSqE[ch], d2);
      }
    }
  } else {
    __syncthreads();
    if (tid < 128) {
      aIg[(size_t)b * JT + qbase + tid] = ldsA[tid] + c0[0];
      aEg[(size_t)b * JT + qbase + tid] = ldsA[128 + tid] + c0[1];
    }
  }
}

// ---------------------------------------------------------------------------
// Branch BN stats -> folded a-coefficients: wfold[c] = w_fus*gamma*rstd,
// c0 = sum_c w_fus*(beta - gamma*rstd*mean).
// ---------------------------------------------------------------------------
__global__ void __launch_bounds__(128) finalize_branches(
    const float* statSumI, const float* statSqI,
    const float* statSumE, const float* statSqE,
    const float* wfus, const float* gI, const float* betI,
    const float* gE, const float* betE,
    float* wfI, float* wfE, float* c0)
{
  __shared__ float r1[128], r2[128];
  int c = threadIdx.x;
  const float invN = 1.0f / (float)NPIX;
  float mI = statSumI[c] * invN;
  float rI = rsqrtf(statSqI[c] * invN - mI * mI + 1e-5f);
  float mE = statSumE[c] * invN;
  float rE = rsqrtf(statSqE[c] * invN - mE * mE + 1e-5f);
  float wi = wfus[c], we = wfus[Cn + c];
  wfI[c] = wi * gI[c] * rI;
  wfE[c] = we * gE[c] * rE;
  r1[c] = wi * (betI[c] - gI[c] * rI * mI);
  r2[c] = we * (betE[c] - gE[c] * rE * mE);
  __syncthreads();
  for (int st = 64; st > 0; st >>= 1) {
    if (c < st) { r1[c] += r1[c + st]; r2[c] += r2[c + st]; }
    __syncthreads();
  }
  if (c == 0) { c0[0] = r1[0]; c0[1] = r2[0]; }
}

// ---------------------------------------------------------------------------
// Per-(b,j) kernel. TDM loads x slab (128c x 256t f32, rows padded 1024B->
// 1040B) into LDS. Computes S,P (rank-1 decomposition of the out-einsum),
// out BN stats, global top-3 of a_int (top-2 of the separable matrix is
// s-independent), and streams path + idx_full outputs.
// ---------------------------------------------------------------------------
__global__ void __launch_bounds__(256) bj_kernel(
    const float* __restrict__ x,
    const float* __restrict__ aIg, const float* __restrict__ aEg,
    const float* __restrict__ bfus,
    float* __restrict__ Sg, float* __restrict__ Pg,
    float* __restrict__ statSumO, float* __restrict__ statSqO,
    float* __restrict__ pathOut, int* __restrict__ idxOut)
{
  extern __shared__ char smemraw[];
  float* xs   = (float*)smemraw;       // 128 x 260 floats (padded rows)
  float* aIv  = xs + 33280;            // 256
  float* aEv  = aIv + 256;             // 256
  int*   top3 = (int*)(aEv + 256);     // 3

  const int tid = threadIdx.x;
  const int wv  = tid >> 5;
  const int b   = blockIdx.x / Jn, j = blockIdx.x % Jn;
  const int bj  = blockIdx.x;

  if (wv == 0) {
    // f32 tensor: dim0=t(256, contiguous), row stride 6400; pad 256DW+4DW
    tdm_load_2d(0, x + (size_t)b * Cn * JT + (size_t)j * Tn,
                /*dsize*/2, /*tdim0*/256, /*tdim1*/128, /*stride0*/JT,
                /*tile0*/256, /*tile1*/128, /*pad*/1, /*256 DW*/7, /*4 DW*/3);
  }
  aIv[tid] = aIg[(size_t)bj * Tn + tid] + bfus[0];
  aEv[tid] = aEg[(size_t)bj * Tn + tid];
  if (wv == 0) __builtin_amdgcn_s_wait_tensorcnt(0);
  __syncthreads();

  // global top-3 of a_int (stable: first index wins on ties, like lax.top_k)
  if (tid == 0) {
    int i0 = 0, i1 = 0, i2 = 0;
    float best = -3.4e38f;
    for (int t = 0; t < Tn; t++) { float v = aIv[t]; if (v > best) { best = v; i0 = t; } }
    best = -3.4e38f;
    for (int t = 0; t < Tn; t++) { if (t == i0) continue; float v = aIv[t]; if (v > best) { best = v; i1 = t; } }
    best = -3.4e38f;
    for (int t = 0; t < Tn; t++) { if (t == i0 || t == i1) continue; float v = aIv[t]; if (v > best) { best = v; i2 = t; } }
    top3[0] = i0; top3[1] = i1; top3[2] = i2;
  }
  __syncthreads();

  // per-s neighbor triple (registers only; thread tid owns s=tid)
  const int s = tid;
  int t0 = top3[0], t1 = top3[1], t2 = top3[2];
  int ca, cb;
  if (t0 == s)      { ca = t1; cb = t2; }
  else if (t1 == s) { ca = t0; cb = t2; }
  else              { ca = t0; cb = t1; }
  int e0 = ca, e1 = cb, e2 = s, tmp;
  if (e0 > e1) { tmp = e0; e0 = e1; e1 = tmp; }
  if (e1 > e2) { tmp = e1; e1 = e2; e2 = tmp; }
  if (e0 > e1) { tmp = e0; e0 = e1; e1 = tmp; }

  // S[c] / P[c]: 2 threads per channel, interleaved t to avoid bank conflicts
  const int c = tid >> 1, h = tid & 1;
  float Sc = 0.f, Pc = 0.f;
  for (int i = 0; i < 128; i++) {
    int t = 2 * i + h;
    float xv = xs[(size_t)c * 260 + t];
    Sc += xv; Pc += aIv[t] * xv;
  }
  Sc += __shfl_xor(Sc, 1, 32);
  Pc += __shfl_xor(Pc, 1, 32);
  if (h == 0) { Sg[(size_t)bj * Cn + c] = Sc; Pg[(size_t)bj * Cn + c] = Pc; }

  // out BN stats over s: out_raw[c,s] = relu(aE[s]*S[c] + P[c])
  float as = 0.f, aq = 0.f;
  for (int i = 0; i < 128; i++) {
    int s2 = h * 128 + i;
    float v = fmaxf(aEv[s2] * Sc + Pc, 0.f);
    as += v; aq += v * v;
  }
  as += __shfl_xor(as, 1, 32);
  aq += __shfl_xor(aq, 1, 32);
  if (h == 0) { atomicAdd(&statSumO[c], as); atomicAdd(&statSqO[c], aq); }

  // stream path (gather x from LDS) and idx_full
  for (int cc = 0; cc < Cn; cc++) {
    float p0 = xs[(size_t)cc * 260 + e0];
    float p1 = xs[(size_t)cc * 260 + e1];
    float p2 = xs[(size_t)cc * 260 + e2];
    size_t off = (((size_t)(b * Cn + cc) * Jn + j) * Tn + s) * 3;
    pathOut[off + 0] = p0; pathOut[off + 1] = p1; pathOut[off + 2] = p2;
    idxOut[off + 0] = e0;  idxOut[off + 1] = e1;  idxOut[off + 2] = e2;
  }
}

__global__ void __launch_bounds__(128) finalize_out(
    const float* sumO, const float* sqO, const float* gO, const float* betO,
    float* scaleO, float* shiftO)
{
  int c = threadIdx.x;
  const float invN = 1.0f / (float)NPIX;
  float m = sumO[c] * invN;
  float r = rsqrtf(sqO[c] * invN - m * m + 1e-5f);
  float sc = gO[c] * r;
  scaleO[c] = sc; shiftO[c] = betO[c] - m * sc;
}

// out[b,c,j,s] = BN(relu(aE[b,j,s]*S[b,j,c] + P[b,j,c]))
__global__ void __launch_bounds__(256) out_write(
    const float* __restrict__ aEg, const float* __restrict__ Sg,
    const float* __restrict__ Pg, const float* __restrict__ scaleO,
    const float* __restrict__ shiftO, float* __restrict__ outp)
{
  int g = blockIdx.x * 256 + threadIdx.x;
  int s = g & 255;
  int rest = g >> 8;           // (b*C + c)*J + j
  int j = rest % Jn;
  int bc = rest / Jn;
  int c = bc & (Cn - 1);
  int b = bc >> 7;
  int bj = b * Jn + j;
  float v = fmaxf(aEg[(size_t)bj * Tn + s] * Sg[(size_t)bj * Cn + c] +
                  Pg[(size_t)bj * Cn + c], 0.f);
  outp[g] = v * scaleO[c] + shiftO[c];
}

// ---------------------------------------------------------------------------
// Host-side orchestration (graph-capture safe: only async ops on `stream`).
// ---------------------------------------------------------------------------
extern "C" void kernel_launch(void* const* d_in, const int* in_sizes, int n_in,
                              void* d_out, int out_size, void* d_ws, size_t ws_size,
                              hipStream_t stream)
{
  const float* x        = (const float*)d_in[0];
  const float* w_int    = (const float*)d_in[1];
  const float* b_int    = (const float*)d_in[2];
  const float* w_ext    = (const float*)d_in[3];
  const float* b_ext    = (const float*)d_in[4];
  const float* w_fus    = (const float*)d_in[5];
  const float* b_fus    = (const float*)d_in[6];
  const float* g_int    = (const float*)d_in[7];
  const float* beta_int = (const float*)d_in[8];
  const float* g_ext    = (const float*)d_in[9];
  const float* beta_ext = (const float*)d_in[10];
  const float* g_out    = (const float*)d_in[11];
  const float* beta_out = (const float*)d_in[12];

  // Workspace layout (~27.5 MB)
  char* ws = (char*)d_ws;
  _Float16* xhT  = (_Float16*)(ws);               // 26,214,400 B
  _Float16* WI16 = (_Float16*)(ws + 26214400);    //     32,768 B
  _Float16* WE16 = (_Float16*)(ws + 26247168);    //     32,768 B
  float* stats   = (float*)(ws + 26279936);       //      3,072 B (6x128)
  float* coefs   = (float*)(ws + 26283008);       //      2,560 B
  float* aIg     = (float*)(ws + 26285568);       //    409,600 B
  float* aEg     = (float*)(ws + 26695168);       //    409,600 B
  float* Sg      = (float*)(ws + 27104768);       //    204,800 B
  float* Pg      = (float*)(ws + 27309568);       //    204,800 B

  float* sumI = stats,       *sqI = stats + 128;
  float* sumE = stats + 256, *sqE = stats + 384;
  float* sumO = stats + 512, *sqO = stats + 640;
  float* wfI = coefs, *wfE = coefs + 128, *c0 = coefs + 256;
  float* scaleO = coefs + 320, *shiftO = coefs + 448;

  // Output layout: out (13,107,200 f32) | path (39,321,600 f32) | idx (39,321,600 i32)
  float* outp  = (float*)d_out;
  float* pathp = outp + (size_t)Bn * Cn * Jn * Tn;
  int*   idxp  = (int*)(pathp + (size_t)Bn * Cn * Jn * Tn * 3);

  (void)hipMemsetAsync(stats, 0, 768 * sizeof(float), stream);

  prep_x<<<51200, 256, 0, stream>>>(x, xhT);
  prep_w<<<64, 256, 0, stream>>>(w_int, w_ext, WI16, WE16);

  const size_t convSh = 34816 + 2 * 128 * sizeof(float);   // padded f16 tile + aI/aE
  conv_wmma_kernel<0><<<800, 256, convSh, stream>>>(
      xhT, WI16, WE16, b_int, b_ext, sumI, sqI, sumE, sqE,
      nullptr, nullptr, nullptr, nullptr, nullptr);
  finalize_branches<<<1, 128, 0, stream>>>(
      sumI, sqI, sumE, sqE, w_fus, g_int, beta_int, g_ext, beta_ext, wfI, wfE, c0);
  conv_wmma_kernel<1><<<800, 256, convSh, stream>>>(
      xhT, WI16, WE16, b_int, b_ext, sumI, sqI, sumE, sqE,
      wfI, wfE, c0, aIg, aEg);

  const size_t bjSh = 33280 * 4 + 256 * 4 + 256 * 4 + 16;  // ~135 KB
  bj_kernel<<<400, 256, bjSh, stream>>>(
      x, aIg, aEg, b_fus, Sg, Pg, sumO, sqO, pathp, idxp);

  finalize_out<<<1, 128, 0, stream>>>(sumO, sqO, g_out, beta_out, scaleO, shiftO);
  out_write<<<51200, 256, 0, stream>>>(aEg, Sg, Pg, scaleO, shiftO, outp);

  (void)in_sizes; (void)n_in; (void)out_size; (void)ws_size;
}
